// LSTMAttentionLM_50002009260560
// MI455X (gfx1250) — matrix operand order repaired and checked by reading, not compiled, not run on hardware
//
#include <hip/hip_runtime.h>

// ---------------- CDNA5 WMMA plumbing (wave32, gfx1250) ----------------
typedef __attribute__((ext_vector_type(16))) __bf16 v16bf;
typedef __attribute__((ext_vector_type(8)))  float  v8f;

// gfx1250 async global->LDS copy path (tracked on ASYNCcnt), if the toolchain
// exposes the builtins; otherwise fall back to regular load/store staging.
#ifdef __has_builtin
#  if __has_builtin(__builtin_amdgcn_global_load_async_to_lds_b128) && \
      __has_builtin(__builtin_amdgcn_s_wait_asynccnt)
#    define USE_ASYNC_LDS 1
#  endif
#endif
#ifndef USE_ASYNC_LDS
#  define USE_ASYNC_LDS 0
#endif

#if USE_ASYNC_LDS
typedef int v4i_vs __attribute__((vector_size(16)));
// GLOBAL_LOAD_ASYNC_TO_LDS_B128: per-lane 16B global -> LDS, ASYNCcnt-tracked
__device__ __forceinline__ void async_cp16(const void* g, void* l) {
    __builtin_amdgcn_global_load_async_to_lds_b128(
        (__attribute__((address_space(1))) v4i_vs*)g,
        (__attribute__((address_space(3))) v4i_vs*)l,
        0, 0);
}
#endif

__device__ __forceinline__ unsigned short f2bf_u(float f) {
    unsigned u = __builtin_bit_cast(unsigned, f);
    u += 0x7FFFu + ((u >> 16) & 1u);          // round-to-nearest-even
    return (unsigned short)(u >> 16);
}
__device__ __forceinline__ void store4bf(unsigned short* d, float4 f) {
    unsigned lo = (unsigned)f2bf_u(f.x) | ((unsigned)f2bf_u(f.y) << 16);
    unsigned hi = (unsigned)f2bf_u(f.z) | ((unsigned)f2bf_u(f.w) << 16);
    *(uint2*)d = make_uint2(lo, hi);
}
// A-matrix 16x32 bf16 fragment: lane half h holds K = h*8+0..7 and 16+h*8+0..7
__device__ __forceinline__ v16bf ldfragA(const unsigned short* p, int half) {
    union { uint4 q[2]; v16bf v; } u;
    u.q[0] = *(const uint4*)(p + half * 8);
    u.q[1] = *(const uint4*)(p + 16 + half * 8);
    return u.v;
}
// B-matrix 32x16 bf16 fragment: lane half h holds K = h*8+0..7 and 16+h*8+0..7
__device__ __forceinline__ v16bf ldfragB(const unsigned short* p, int half) {
    union { uint4 q[2]; v16bf v; } u;
    u.q[0] = *(const uint4*)(p + half * 8);
    u.q[1] = *(const uint4*)(p + 16 + half * 8);
    return u.v;
}
__device__ __forceinline__ float sigf(float x) { return 1.f / (1.f + __expf(-x)); }

// ---------------- generic WMMA GEMM: C = ep(A[M,K] @ W[N,K]^T + bias (+adds)) ----
// block tile 64(M) x 128(N), 8 waves, KB=32, bf16 staged in LDS (pad stride 40)
template<bool HAS_BIAS, bool HAS_ADDFULL, bool HAS_ADDROW, bool DO_TANH>
__global__ __launch_bounds__(256)
void gemm_kernel(const float* __restrict__ A, int lda,
                 const float* __restrict__ W, int ldw,
                 const float* __restrict__ bias,
                 const float* __restrict__ addfull,
                 const float* __restrict__ addrow, int rowsPerBatch,
                 float* __restrict__ C, int N, int K)
{
    __shared__ unsigned short As[64 * 40];
    __shared__ unsigned short Bs[128 * 40];
    const int tid  = threadIdx.x;
    const int wave = tid >> 5, lane = tid & 31;
    const int half = lane >> 4, l16 = lane & 15;
    const int mt   = wave & 3;          // m-tile 0..3
    const int ng2  = wave >> 2;         // n-group 0..1 (4 n-tiles each)
    const int row0 = blockIdx.y * 64;
    const int col0 = blockIdx.x * 128;

    v8f acc[4] = {};
    for (int k0 = 0; k0 < K; k0 += 32) {
        // stage A tile 64x32 (f32 -> bf16)
        for (int i = tid; i < 512; i += 256) {
            int r = i >> 3, c4 = i & 7;
            float4 f = *(const float4*)(A + (size_t)(row0 + r) * lda + k0 + c4 * 4);
            store4bf(&As[r * 40 + c4 * 4], f);
        }
        // stage B tile 128x32 from weight rows (f32 -> bf16), stored [n][k]
        for (int i = tid; i < 1024; i += 256) {
            int r = i >> 3, c4 = i & 7;
            float4 f = *(const float4*)(W + (size_t)(col0 + r) * ldw + k0 + c4 * 4);
            store4bf(&Bs[r * 40 + c4 * 4], f);
        }
        if (k0 + 32 < K) {   // hint next k-slab into the cache hierarchy
            __builtin_prefetch(A + (size_t)(row0 + (tid >> 2)) * lda + k0 + 32, 0, 1);
            __builtin_prefetch(W + (size_t)(col0 + (tid >> 1)) * ldw + k0 + 32, 0, 1);
        }
        __syncthreads();
        v16bf a = ldfragA(&As[(mt * 16 + l16) * 40], half);
#pragma unroll
        for (int j = 0; j < 4; ++j) {
            v16bf b = ldfragB(&Bs[((ng2 * 4 + j) * 16 + l16) * 40], half);
            acc[j] = __builtin_amdgcn_wmma_f32_16x16x32_bf16(
                         false, a, false, b, (short)0, acc[j], false, false);
        }
        __syncthreads();
    }
    // epilogue: lane holds C[m = vi+8*half][n = l16] per ISA C/D layout
    const int bidx = HAS_ADDROW ? (row0 / rowsPerBatch) : 0;
#pragma unroll
    for (int j = 0; j < 4; ++j) {
        const int cg = col0 + (ng2 * 4 + j) * 16 + l16;
        float badd = 0.f;
        if (HAS_BIAS)   badd += bias[cg];
        if (HAS_ADDROW) badd += addrow[(size_t)bidx * N + cg];
#pragma unroll
        for (int vi = 0; vi < 8; ++vi) {
            const int rg = row0 + mt * 16 + vi + 8 * half;
            float x = acc[j][vi] + badd;
            if (HAS_ADDFULL) x += addfull[(size_t)rg * N + cg];
            if (DO_TANH)     x = tanhf(x);
            C[(size_t)rg * N + cg] = x;
        }
    }
}

// ---------------- cooperative LSTM recurrence -------------------------
// 16 WGs x 256 thr; WG w owns h columns [32w,32w+32). Per step:
//   gates(16x2048) = x_pre[:,t,:] + h @ Whh^T  via 8 WMMA tiles per WG,
//   W_hh slice (bf16) streamed from WGP$/L2 (compiler hoists the 16
//   loop-invariant B-fragments into VGPRs), h broadcast through a global
//   bf16 ping-pong buffer (async-DMA'd into LDS), device-scope atomic
//   counter as per-step grid barrier.
__global__ __launch_bounds__(256)
void lstm_kernel(const float* __restrict__ xpre,          // (B,S,4H)
                 const unsigned short* __restrict__ Whhb, // (4H,H) bf16
                 float* __restrict__ out,                  // (B,S,H)
                 unsigned short* __restrict__ hbuf,        // (2,B,H) bf16 ping-pong
                 float* __restrict__ hN, float* __restrict__ cN, // (B,H)
                 unsigned int* __restrict__ counter)
{
    __shared__ unsigned short hl[16 * 520];      // h tile (pad stride 520)
    __shared__ float gbuf[8][16][16];            // 8 gate tiles
    __shared__ float cbuf[16][32];               // persistent cell state slice
    const int w    = blockIdx.x;
    const int tid  = threadIdx.x;
    const int wave = tid >> 5, lane = tid & 31;
    const int half = lane >> 4, l16 = lane & 15;
    const int gi   = wave >> 1, ct = wave & 1;   // gate idx, col-tile within WG

    for (int i = tid; i < 512; i += 256) cbuf[i >> 5][i & 31] = 0.f;

    for (int t = 0; t < 512; ++t) {
        const unsigned short* hg = hbuf + (size_t)(t & 1) * 8192;
#if USE_ASYNC_LDS
        for (int i = tid; i < 1024; i += 256) {  // 16x512 bf16 -> LDS, async DMA
            int b = i >> 6, c8 = i & 63;
            async_cp16(hg + b * 512 + c8 * 8, &hl[b * 520 + c8 * 8]);
        }
        __builtin_amdgcn_s_wait_asynccnt(0);
#else
        for (int i = tid; i < 1024; i += 256) {  // 16x512 bf16 -> LDS (uint4)
            int b = i >> 6, c8 = i & 63;
            *(uint4*)(&hl[b * 520 + c8 * 8]) = *(const uint4*)(hg + b * 512 + c8 * 8);
        }
#endif
        __syncthreads();

        v8f acc = {};
        const unsigned short* ab = &hl[l16 * 520];
        const unsigned short* bb = Whhb + (size_t)(gi * 512 + w * 32 + ct * 16 + l16) * 512;
        v16bf a_cur = ldfragA(ab, half);         // software-pipeline A-frags so the
#pragma unroll                                   // next ds_load pair overlaps the wmma
        for (int kk = 0; kk < 16; ++kk) {
            v16bf a_nxt = a_cur;
            if (kk < 15) a_nxt = ldfragA(ab + (kk + 1) * 32, half);
            v16bf b = ldfragB(bb + kk * 32, half);
            acc = __builtin_amdgcn_wmma_f32_16x16x32_bf16(
                      false, a_cur, false, b, (short)0, acc, false, false);
            a_cur = a_nxt;
        }
#pragma unroll
        for (int vi = 0; vi < 8; ++vi) gbuf[wave][vi + 8 * half][l16] = acc[vi];
        __syncthreads();

        for (int e = tid; e < 512; e += 256) {   // 16 batches x 32 owned cols
            int b = e >> 5, cl = e & 31;
            int ctl = cl >> 4, c16 = cl & 15;
            size_t xb = ((size_t)(b * 512 + t)) * 2048 + w * 32 + cl;
            float g_i = gbuf[0 * 2 + ctl][b][c16] + xpre[xb];
            float g_f = gbuf[1 * 2 + ctl][b][c16] + xpre[xb + 512];
            float g_g = gbuf[2 * 2 + ctl][b][c16] + xpre[xb + 1024];
            float g_o = gbuf[3 * 2 + ctl][b][c16] + xpre[xb + 1536];
            float cold = cbuf[b][cl];
            float cn   = sigf(g_f) * cold + sigf(g_i) * tanhf(g_g);
            float hn   = sigf(g_o) * tanhf(cn);
            cbuf[b][cl] = cn;
            out[((size_t)(b * 512 + t)) * 512 + w * 32 + cl] = hn;
            hbuf[(size_t)((t + 1) & 1) * 8192 + b * 512 + w * 32 + cl] = f2bf_u(hn);
            if (t == 511) {
                hN[b * 512 + w * 32 + cl] = hn;
                cN[b * 512 + w * 32 + cl] = cn;
            }
        }
        __threadfence();
        __syncthreads();
        if (tid == 0) {
            __hip_atomic_fetch_add(counter, 1u, __ATOMIC_RELEASE, __HIP_MEMORY_SCOPE_AGENT);
            unsigned target = 16u * (unsigned)(t + 1);
            while (__hip_atomic_load(counter, __ATOMIC_ACQUIRE, __HIP_MEMORY_SCOPE_AGENT) < target)
                __builtin_amdgcn_s_sleep(1);
        }
        __syncthreads();
    }
}

// ---------------- small helper kernels ---------------------------------
__global__ void bias_sum_kernel(const float* bi0, const float* bh0,
                                const float* bi1, const float* bh1,
                                float* s0, float* s1) {
    int i = blockIdx.x * 256 + threadIdx.x;         // 2048
    s0[i] = bi0[i] + bh0[i];
    s1[i] = bi1[i] + bh1[i];
}

__global__ void cvt_whh_kernel(const float* w0, const float* w1,
                               unsigned short* d0, unsigned short* d1) {
    int i = blockIdx.x * 256 + threadIdx.x;         // 262144 float4 each
    float4 f = *(const float4*)(w0 + (size_t)i * 4); store4bf(d0 + (size_t)i * 4, f);
    f = *(const float4*)(w1 + (size_t)i * 4);        store4bf(d1 + (size_t)i * 4, f);
}

__global__ void embed_kernel(const int* __restrict__ x, const float* __restrict__ emb,
                             float* __restrict__ e) {
    int idx = blockIdx.x * 256 + threadIdx.x;       // 524288 float4
    int row = idx >> 6, c4 = idx & 63;
    int tok = x[row];
    *(float4*)(e + (size_t)row * 256 + c4 * 4) =
        *(const float4*)(emb + (size_t)tok * 256 + c4 * 4);
}

// qa[b,n] = dot(out1[b,S-1,:], Wa[n, 0:H])
__global__ void qa_kernel(const float* __restrict__ out1, const float* __restrict__ Wa,
                          float* __restrict__ qa) {
    int b = blockIdx.y;
    int n = blockIdx.x * 256 + threadIdx.x;
    const float* q = out1 + ((size_t)(b * 512 + 511)) * 512;
    const float* wr = Wa + (size_t)n * 1024;
    float s = 0.f;
    for (int k = 0; k < 512; ++k) s += q[k] * wr[k];
    qa[b * 512 + n] = s;
}

// scores[r] = dot(energy[r,:], v)   — one wave per row
__global__ void scores_kernel(const float* __restrict__ energy, const float* __restrict__ v,
                              float* __restrict__ scores) {
    int wave = threadIdx.x >> 5, lane = threadIdx.x & 31;
    int row = blockIdx.x * 8 + wave;
    const float* er = energy + (size_t)row * 512;
    float s = 0.f;
    for (int k = lane; k < 512; k += 32) s += er[k] * v[k];
#pragma unroll
    for (int off = 16; off; off >>= 1) s += __shfl_xor(s, off, 32);
    if (lane == 0) scores[row] = s;
}

// causal softmax over position-only scores -> prefix-sum context
__global__ void attn_kernel(const float* __restrict__ scores, const float* __restrict__ out1,
                            float* __restrict__ context) {
    __shared__ float sw[512];
    __shared__ float zp[512];
    __shared__ float red[512];
    int b = blockIdx.x, tx = threadIdx.x;
    float s = scores[b * 512 + tx];
    red[tx] = s; __syncthreads();
    for (int off = 256; off; off >>= 1) {
        if (tx < off) red[tx] = fmaxf(red[tx], red[tx + off]);
        __syncthreads();
    }
    sw[tx] = __expf(s - red[0]);
    __syncthreads();
    if (tx == 0) { float r = 0.f; for (int j = 0; j < 512; ++j) { r += sw[j]; zp[j] = r; } }
    __syncthreads();
    float p = 0.f;
    for (int t = 0; t < 512; ++t) {
        p += sw[t] * out1[((size_t)(b * 512 + t)) * 512 + tx];
        context[((size_t)(b * 512 + t)) * 512 + tx] = p / zp[t];
    }
}

// t=0 bypasses attention: attended[:,0,:] = lstm_out[:,0,:]
__global__ void fix_row0_kernel(const float* __restrict__ out1, float* __restrict__ combined) {
    int i = blockIdx.x * 256 + threadIdx.x;   // 8192
    int b = i >> 9, h = i & 511;
    combined[((size_t)b * 512) * 512 + h] = out1[((size_t)b * 512) * 512 + h];
}

// ---------------- launcher ----------------------------------------------
extern "C" void kernel_launch(void* const* d_in, const int* in_sizes, int n_in,
                              void* d_out, int out_size, void* d_ws, size_t ws_size,
                              hipStream_t stream) {
    (void)in_sizes; (void)n_in; (void)out_size; (void)ws_size;
    const float* emb  = (const float*)d_in[1];
    const int*   x    = (const int*)d_in[0];
    const float* Wih0 = (const float*)d_in[2];
    const float* Whh0 = (const float*)d_in[3];
    const float* bih0 = (const float*)d_in[4];
    const float* bhh0 = (const float*)d_in[5];
    const float* Wih1 = (const float*)d_in[6];
    const float* Whh1 = (const float*)d_in[7];
    const float* bih1 = (const float*)d_in[8];
    const float* bhh1 = (const float*)d_in[9];
    const float* Wa   = (const float*)d_in[10];
    const float* ba   = (const float*)d_in[11];
    const float* vv   = (const float*)d_in[12];
    const float* Wc   = (const float*)d_in[13];
    const float* bc   = (const float*)d_in[14];
    const float* Wfc  = (const float*)d_in[15];
    const float* bfc  = (const float*)d_in[16];

    float* w = (float*)d_ws;
    float* e        = w;                              // 8192*256
    float* xpre     = e        + 2097152;             // 8192*2048 (shared L0/L1)
    float* out0     = xpre     + 16777216;            // 8192*512
    float* out1     = out0     + 4194304;
    float* energy   = out1     + 4194304;             // also reused as tmpc
    float* context  = energy   + 4194304;
    float* combined = context  + 4194304;
    float* qa       = combined + 4194304;             // 16*512
    float* scores   = qa       + 8192;                // 16*512
    float* bsum0    = scores   + 8192;                // 2048
    float* bsum1    = bsum0    + 2048;                // 2048
    unsigned short* whh0b = (unsigned short*)(bsum1 + 2048);  // 2048*512 bf16
    unsigned short* whh1b = whh0b + 1048576;
    unsigned short* hbuf  = whh1b + 1048576;                  // 2*16*512 bf16
    unsigned int*   cnt   = (unsigned int*)(hbuf + 16384);    // [0]=L0, [1]=L1

    float* logits = (float*)d_out;                    // 8192*32000
    float* hc     = logits + (size_t)8192 * 32000;    // h_n then c_n
    float* hN0 = hc;         float* hN1 = hc + 8192;
    float* cN0 = hc + 16384; float* cN1 = hc + 24576;

    (void)hipMemsetAsync(cnt, 0, 8, stream);
    bias_sum_kernel<<<8, 256, 0, stream>>>(bih0, bhh0, bih1, bhh1, bsum0, bsum1);
    cvt_whh_kernel<<<1024, 256, 0, stream>>>(Whh0, Whh1, whh0b, whh1b);
    embed_kernel<<<2048, 256, 0, stream>>>(x, emb, e);

    // x_pre0 = e @ Wih0^T + (bih0+bhh0)
    gemm_kernel<true, false, false, false><<<dim3(16, 128), 256, 0, stream>>>(
        e, 256, Wih0, 256, bsum0, nullptr, nullptr, 512, xpre, 2048, 256);
    (void)hipMemsetAsync(hbuf, 0, 2 * 8192 * sizeof(unsigned short), stream);
    lstm_kernel<<<16, 256, 0, stream>>>(xpre, whh0b, out0, hbuf, hN0, cN0, cnt + 0);

    // x_pre1 = out0 @ Wih1^T + (bih1+bhh1)
    gemm_kernel<true, false, false, false><<<dim3(16, 128), 256, 0, stream>>>(
        out0, 512, Wih1, 512, bsum1, nullptr, nullptr, 512, xpre, 2048, 512);
    (void)hipMemsetAsync(hbuf, 0, 2 * 8192 * sizeof(unsigned short), stream);
    lstm_kernel<<<16, 256, 0, stream>>>(xpre, whh1b, out1, hbuf, hN1, cN1, cnt + 1);

    // attention
    qa_kernel<<<dim3(2, 16), 256, 0, stream>>>(out1, Wa, qa);
    gemm_kernel<true, false, true, true><<<dim3(4, 128), 256, 0, stream>>>(
        out1, 512, Wa + 512, 1024, ba, nullptr, qa, 512, energy, 512, 512);
    scores_kernel<<<1024, 256, 0, stream>>>(energy, vv, scores);
    attn_kernel<<<16, 512, 0, stream>>>(scores, out1, context);

    // combined = tanh(out1 @ Wc[:, :H]^T + context @ Wc[:, H:]^T + bc)
    gemm_kernel<false, false, false, false><<<dim3(4, 128), 256, 0, stream>>>(
        out1, 512, Wc, 1024, nullptr, nullptr, nullptr, 512, energy /*tmpc*/, 512, 512);
    gemm_kernel<true, true, false, true><<<dim3(4, 128), 256, 0, stream>>>(
        context, 512, Wc + 512, 1024, bc, energy /*tmpc*/, nullptr, 512, combined, 512, 512);
    fix_row0_kernel<<<32, 256, 0, stream>>>(out1, combined);

    // logits = attended @ Wfc^T + bfc   (dominant GEMM: 8192 x 32000 x 512)
    gemm_kernel<true, false, false, false><<<dim3(250, 128), 256, 0, stream>>>(
        combined, 512, Wfc, 512, bfc, nullptr, nullptr, 512, logits, 32000, 512);
}
